// SpanMarkerV0_36017595744715
// MI455X (gfx1250) — matrix operand order repaired and checked by reading, not compiled, not run on hardware
//
#include <hip/hip_runtime.h>

typedef __attribute__((ext_vector_type(16))) _Float16 v16h;
typedef __attribute__((ext_vector_type(8)))  _Float16 v8h;
typedef __attribute__((ext_vector_type(4)))  _Float16 v4h;
typedef __attribute__((ext_vector_type(8)))  float    v8f;
typedef __attribute__((ext_vector_type(4)))  int      v4i;

#define ASG __attribute__((address_space(1)))
#define ASL __attribute__((address_space(3)))

#if __has_builtin(__builtin_amdgcn_global_load_async_to_lds_b128)
#define ASYNC_LDS 1
#else
#define ASYNC_LDS 0
#endif

constexpr int Dd   = 1024;
constexpr int Bb   = 4;
constexpr int Ll   = 4096;
constexpr int Ss   = 2048;
constexpr int ROWS = Bb * Ss;   // 8192 gathered rows (= M for every GEMM)

constexpr int LDK  = 40;        // padded K-stride in LDS (breaks bank conflicts)

__device__ __forceinline__ void wait_async0() {
#if ASYNC_LDS
#if __has_builtin(__builtin_amdgcn_s_wait_asynccnt)
  __builtin_amdgcn_s_wait_asynccnt(0);
#else
  asm volatile("s_wait_asynccnt 0x0" ::: "memory");
#endif
#endif
}

// ---------------------------------------------------------------------------
// Gather h rows at clipped span indices, convert f32 -> f16.
// One block per (b,s) pair; 256 threads x float4 covers D=1024 exactly.
// ---------------------------------------------------------------------------
__global__ __launch_bounds__(256) void gather_convert_f16(
    const float* __restrict__ h, const int* __restrict__ span,
    _Float16* __restrict__ Xs, _Float16* __restrict__ Xe)
{
  const int row = blockIdx.x;          // 0..ROWS-1
  const int b   = row / Ss;
  int i0 = span[2 * row + 0];
  int i1 = span[2 * row + 1];
  i0 = i0 < 0 ? 0 : (i0 > Ll - 1 ? Ll - 1 : i0);
  i1 = i1 < 0 ? 0 : (i1 > Ll - 1 ? Ll - 1 : i1);

  const float4* s0 = (const float4*)(h + ((size_t)b * Ll + i0) * Dd);
  const float4* s1 = (const float4*)(h + ((size_t)b * Ll + i1) * Dd);
  const int t = threadIdx.x;           // 0..255, 4 floats each

  float4 a = s0[t];
  float4 c = s1[t];
  v4h pa = { (_Float16)a.x, (_Float16)a.y, (_Float16)a.z, (_Float16)a.w };
  v4h pc = { (_Float16)c.x, (_Float16)c.y, (_Float16)c.z, (_Float16)c.w };
  *(v4h*)&Xs[(size_t)row * Dd + 4 * t] = pa;
  *(v4h*)&Xe[(size_t)row * Dd + 4 * t] = pc;
}

// ---------------------------------------------------------------------------
// C = act(A (MxK, f16 row-major) * W^T (W is NxK f32 row-major) + bias)
// mode==1: relu, store f16 at ldc/c_off.  mode==0: no relu, store f32.
// Block tile 256x128, 8 waves (4x2), wave tile 64x64 = 4x4 WMMA 16x16x32.
// Double-buffered LDS; A tile staged via async-load-to-LDS (CDNA5 path),
// B tile converted f32->f16 through VGPRs overlapping the WMMAs.
// All dims are multiples of the tile sizes for this problem.
// ---------------------------------------------------------------------------
__global__ __launch_bounds__(256) void gemm_wmma_bias_act(
    const _Float16* __restrict__ A, int lda,
    const float* __restrict__ W, int ldw,
    const float* __restrict__ bias,
    void* __restrict__ Cout, int ldc, int c_off,
    int K, int mode)
{
  __shared__ __align__(32) _Float16 As[2][256][LDK];  // 40 KB
  __shared__ __align__(32) _Float16 Bs[2][128][LDK];  // 20 KB

  const int tid  = threadIdx.x;
  const int lane = tid & 31;
  const int wave = tid >> 5;
  const int wm   = (wave >> 1) * 64;   // wave row offset in block tile (0..192)
  const int wn   = (wave & 1) * 64;    // wave col offset in block tile (0/64)
  const int bm   = blockIdx.y * 256;
  const int bn   = blockIdx.x * 128;

  const v8f zero = {};
  v8f acc[4][4];
#pragma unroll
  for (int i = 0; i < 4; ++i)
#pragma unroll
    for (int j = 0; j < 4; ++j) acc[i][j] = zero;

  // ---- staging mappings ----
  // A: 256 rows x 32 halfs; thread t stages row t as 4x16B segments.
  const _Float16* Agr = A + (size_t)(bm + tid) * lda;
  // B: 128 rows x 32 f32; 2 threads per row, 16 floats each.
  const int sr = tid >> 1;
  const int sc = (tid & 1) << 4;
  const float* Wgr = W + (size_t)(bn + sr) * ldw + sc;

  // ---- fragment addressing (ISA wave32 16-bit A 16x32 / B 32x16 layouts) ----
  const int am = lane & 15;
  const int ak = (lane >> 4) << 3;     // 0 or 8
  const int bl = lane & 15;
  const int bk = (lane >> 4) << 3;     // 0 or 8

  auto stageA = [&](int kk, int buf) {
    const _Float16* gp = Agr + kk;
#pragma unroll
    for (int s = 0; s < 4; ++s) {
#if ASYNC_LDS
      __builtin_amdgcn_global_load_async_to_lds_b128(
          (ASG v4i*)(gp + s * 8), (ASL v4i*)&As[buf][tid][s * 8], 0, 0);
#else
      *(v8h*)&As[buf][tid][s * 8] = *(const v8h*)(gp + s * 8);
#endif
    }
  };
  auto loadB = [&](int kk, float4* w) {
    const float* p = Wgr + kk;
    w[0] = *(const float4*)(p);
    w[1] = *(const float4*)(p + 4);
    w[2] = *(const float4*)(p + 8);
    w[3] = *(const float4*)(p + 12);
  };
  auto storeB = [&](int buf, const float4* w) {
    v8h p0 = { (_Float16)w[0].x, (_Float16)w[0].y, (_Float16)w[0].z, (_Float16)w[0].w,
               (_Float16)w[1].x, (_Float16)w[1].y, (_Float16)w[1].z, (_Float16)w[1].w };
    v8h p1 = { (_Float16)w[2].x, (_Float16)w[2].y, (_Float16)w[2].z, (_Float16)w[2].w,
               (_Float16)w[3].x, (_Float16)w[3].y, (_Float16)w[3].z, (_Float16)w[3].w };
    *(v8h*)&Bs[buf][sr][sc]     = p0;
    *(v8h*)&Bs[buf][sr][sc + 8] = p1;
  };

  // ---- prologue: stage k=0 into buffer 0 ----
  {
    float4 w[4];
    stageA(0, 0);
    loadB(0, w);
    storeB(0, w);
  }
  wait_async0();
  __syncthreads();

  // ---- main loop: compute buffer p while staging buffer p^1 ----
  for (int k0 = 0; k0 < K; k0 += 32) {
    const int  p    = (k0 >> 5) & 1;
    const bool more = (k0 + 32) < K;

    float4 w[4];
    if (more) {
      stageA(k0 + 32, p ^ 1);          // async: in flight during WMMAs
      loadB(k0 + 32, w);               // f32 weight loads in flight too
    }

    // fragments from buffer p
    v16h af[4];
#pragma unroll
    for (int i = 0; i < 4; ++i) {
      const _Float16* ap = &As[p][wm + i * 16 + am][0];
      v8h lo = *(const v8h*)(ap + ak);
      v8h hi = *(const v8h*)(ap + ak + 16);
      af[i] = __builtin_shufflevector(lo, hi, 0, 1, 2, 3, 4, 5, 6, 7,
                                      8, 9, 10, 11, 12, 13, 14, 15);
    }
    v16h bf[4];
#pragma unroll
    for (int j = 0; j < 4; ++j) {
      const _Float16* bp = &Bs[p][wn + j * 16 + bl][0];
      v8h lo = *(const v8h*)(bp + bk);
      v8h hi = *(const v8h*)(bp + bk + 16);
      bf[j] = __builtin_shufflevector(lo, hi, 0, 1, 2, 3, 4, 5, 6, 7,
                                      8, 9, 10, 11, 12, 13, 14, 15);
    }

    // 16 WMMAs per wave per K-step
#pragma unroll
    for (int i = 0; i < 4; ++i)
#pragma unroll
      for (int j = 0; j < 4; ++j)
        acc[i][j] = __builtin_amdgcn_wmma_f32_16x16x32_f16(
            false, af[i], false, bf[j], (short)0, acc[i][j], false, false);

    if (more) storeB(p ^ 1, w);        // convert + ds_store overlaps tail
    wait_async0();
    __syncthreads();
  }

  // ---- epilogue: bias + activation + store ----
  // C 16x16 f32 layout: VGPR r, lanes 0-15 -> M=r, lanes 16-31 -> M=8+r; N=lane&15
  const int colb = bn + wn + (lane & 15);
  const int rowb = bm + wm + ((lane >> 4) << 3);
  if (mode) {
    _Float16* Cp = (_Float16*)Cout;
#pragma unroll
    for (int i = 0; i < 4; ++i) {
#pragma unroll
      for (int j = 0; j < 4; ++j) {
        const int col = colb + j * 16;
        const float bv = bias[col];
#pragma unroll
        for (int r = 0; r < 8; ++r) {
          float v = acc[i][j][r] + bv;
          v = v > 0.f ? v : 0.f;
          Cp[(size_t)(rowb + i * 16 + r) * ldc + c_off + col] = (_Float16)v;
        }
      }
    }
  } else {
    float* Cp = (float*)Cout;
#pragma unroll
    for (int i = 0; i < 4; ++i) {
#pragma unroll
      for (int j = 0; j < 4; ++j) {
        const int col = colb + j * 16;
        const float bv = bias[col];
#pragma unroll
        for (int r = 0; r < 8; ++r) {
          Cp[(size_t)(rowb + i * 16 + r) * ldc + c_off + col] = acc[i][j][r] + bv;
        }
      }
    }
  }
}

// ---------------------------------------------------------------------------
extern "C" void kernel_launch(void* const* d_in, const int* in_sizes, int n_in,
                              void* d_out, int out_size, void* d_ws, size_t ws_size,
                              hipStream_t stream) {
  (void)in_sizes; (void)n_in; (void)out_size; (void)ws_size;

  const float* h     = (const float*)d_in[0];
  const int*   span  = (const int*)d_in[1];
  const float* ps_w1 = (const float*)d_in[2];
  const float* ps_b1 = (const float*)d_in[3];
  const float* ps_w2 = (const float*)d_in[4];
  const float* ps_b2 = (const float*)d_in[5];
  const float* pe_w1 = (const float*)d_in[6];
  const float* pe_b1 = (const float*)d_in[7];
  const float* pe_w2 = (const float*)d_in[8];
  const float* pe_b2 = (const float*)d_in[9];
  const float* op_w1 = (const float*)d_in[10];
  const float* op_b1 = (const float*)d_in[11];
  const float* op_w2 = (const float*)d_in[12];
  const float* op_b2 = (const float*)d_in[13];

  // workspace layout (f16): Xs 8192x1024 | Xe 8192x1024 | H 8192x4096 | Cat 8192x2048
  _Float16* Xs  = (_Float16*)d_ws;
  _Float16* Xe  = Xs  + (size_t)ROWS * Dd;
  _Float16* H   = Xe  + (size_t)ROWS * Dd;
  _Float16* Cat = H   + (size_t)ROWS * 4 * Dd;     // total 128 MB

  // 1) gather rows at span start/end indices, convert to f16
  gather_convert_f16<<<ROWS, 256, 0, stream>>>(h, span, Xs, Xe);

  const dim3 blk(256);
  const dim3 gN4096(4096 / 128, ROWS / 256);   // N=4096 -> (32, 32)
  const dim3 gN1024(1024 / 128, ROWS / 256);   // N=1024 -> (8, 32)

  // 2) H = relu(Xs * ps_w1^T + ps_b1)            (8192x4096)
  gemm_wmma_bias_act<<<gN4096, blk, 0, stream>>>(Xs, Dd, ps_w1, Dd, ps_b1,
                                                 H, 4 * Dd, 0, Dd, 1);
  // 3) Cat[:, :1024] = relu(H * ps_w2^T + ps_b2) (start span rep, relu folded)
  gemm_wmma_bias_act<<<gN1024, blk, 0, stream>>>(H, 4 * Dd, ps_w2, 4 * Dd, ps_b2,
                                                 Cat, 2 * Dd, 0, 4 * Dd, 1);
  // 4) H = relu(Xe * pe_w1^T + pe_b1)
  gemm_wmma_bias_act<<<gN4096, blk, 0, stream>>>(Xe, Dd, pe_w1, Dd, pe_b1,
                                                 H, 4 * Dd, 0, Dd, 1);
  // 5) Cat[:, 1024:] = relu(H * pe_w2^T + pe_b2) (end span rep, relu folded)
  gemm_wmma_bias_act<<<gN1024, blk, 0, stream>>>(H, 4 * Dd, pe_w2, 4 * Dd, pe_b2,
                                                 Cat, 2 * Dd, Dd, 4 * Dd, 1);
  // 6) H = relu(Cat * op_w1^T + op_b1)           (8192x4096, K=2048)
  gemm_wmma_bias_act<<<gN4096, blk, 0, stream>>>(Cat, 2 * Dd, op_w1, 2 * Dd, op_b1,
                                                 H, 4 * Dd, 0, 2 * Dd, 1);
  // 7) out = H * op_w2^T + op_b2                 (f32, no relu)
  gemm_wmma_bias_act<<<gN1024, blk, 0, stream>>>(H, 4 * Dd, op_w2, 4 * Dd, op_b2,
                                                 d_out, Dd, 0, 4 * Dd, 0);
}
